// ResidualLSTMCell_64518998720971
// MI455X (gfx1250) — matrix operand order repaired and checked by reading, not compiled, not run on hardware
//
#include <hip/hip_runtime.h>

// ---------------------------------------------------------------------------
// Types for CDNA5 WMMA (wave32): V_WMMA_F32_16X16X32_BF16
// ---------------------------------------------------------------------------
typedef __attribute__((ext_vector_type(16))) __bf16 v16bf;
typedef __attribute__((ext_vector_type(8)))  __bf16 v8bf;
typedef __attribute__((ext_vector_type(4)))  __bf16 v4bf;
typedef __attribute__((ext_vector_type(8)))  float  v8f;

struct V16Pair { v8bf lo, hi; };
static __device__ __forceinline__ v16bf mk16(v8bf lo, v8bf hi) {
    V16Pair p{lo, hi};
    return __builtin_bit_cast(v16bf, p);
}

// fp32 -> bf16, round-to-nearest-even
static __device__ __forceinline__ __bf16 f2bf(float x) {
    unsigned u = __builtin_bit_cast(unsigned, x);
    u = (u + 0x7FFFu + ((u >> 16) & 1u)) >> 16;
    unsigned short s = (unsigned short)u;
    return __builtin_bit_cast(__bf16, s);
}

static __device__ __forceinline__ float sigmoidf_(float x) {
    return 1.0f / (1.0f + __expf(-x));
}

// LDS byte-offset of a pointer into a __shared__ array (AS(3) pointer value
// == LDS offset; generic -> AS(3) is an addrspacecast).
typedef __attribute__((address_space(3))) __bf16 lds_bf16_t;
static __device__ __forceinline__ unsigned lds_off(__bf16* p) {
    return (unsigned)(unsigned long long)(lds_bf16_t*)p;
}

// ---------------------------------------------------------------------------
// Problem constants
// ---------------------------------------------------------------------------
#define BB 4096            // batch
#define HH 2048            // hidden (= D)
#define NE (2048u * 2048u) // elements per weight matrix
#define ACT ((size_t)BB * HH)
#define LDZ (5 * HH)       // Z is [B, 5H]

// ---------------------------------------------------------------------------
// Convert fp32 -> bf16 (activations), vectorized 4-wide
// ---------------------------------------------------------------------------
__global__ void __launch_bounds__(256)
conv_bf16_kernel(const float* __restrict__ src, __bf16* __restrict__ dst, int n4) {
    int i = blockIdx.x * blockDim.x + threadIdx.x;
    if (i < n4) {
        float4 v = ((const float4*)src)[i];
        v4bf o;
        o[0] = f2bf(v.x); o[1] = f2bf(v.y); o[2] = f2bf(v.z); o[3] = f2bf(v.w);
        ((v4bf*)dst)[i] = o;
    }
}

// ---------------------------------------------------------------------------
// Convert fp32 -> bf16 with transpose (weights): dst[n][k] = bf16(src[k][n])
// 2048 x 2048 only. block (32,8), grid (64,64).
// ---------------------------------------------------------------------------
__global__ void __launch_bounds__(256)
convT_bf16_kernel(const float* __restrict__ src, __bf16* __restrict__ dst) {
    __shared__ __bf16 t[32][34];
    const int bx = blockIdx.x * 32, by = blockIdx.y * 32;
    const int tx = threadIdx.x, ty = threadIdx.y;
#pragma unroll
    for (int r = 0; r < 32; r += 8)
        t[ty + r][tx] = f2bf(src[(size_t)(by + ty + r) * HH + bx + tx]);
    __syncthreads();
#pragma unroll
    for (int r = 0; r < 32; r += 8)
        dst[(size_t)(bx + ty + r) * HH + by + tx] = t[tx][ty + r];
}

// ---------------------------------------------------------------------------
// Tiled WMMA GEMM:  C[m,n] (+)= sum_s A_s[m,:] . W_s[:,n]
//   A_s : [4096 x 2048] bf16 row-major (batch-major)
//   Bt_s: [2048 x 2048] bf16, stored N-major (i.e. W^T), row n = output col
//   C   : fp32, leading dim ldc; BETA=1 -> accumulate into existing C
// BM=BN=128, BK=32; 256 threads = 8 waves; wave tile 32x64 (2x4 WMMA frags).
// Staging: GLOBAL_LOAD_ASYNC_TO_LDS_B128 (ASYNCcnt), 3-deep pipeline
// (prefetch distance 2, 8 DMAs in flight, in-order completion => wait <= 4).
// ---------------------------------------------------------------------------
#define PITCH 40   // LDS row pitch in bf16 elements (80B, 16B-aligned, 20-bank stride)

template <int BETA>
__global__ void __launch_bounds__(256)
wmma_gemm_bf16(const __bf16* __restrict__ A0, const __bf16* __restrict__ Bt0,
               const __bf16* __restrict__ A1, const __bf16* __restrict__ Bt1,
               const __bf16* __restrict__ A2, const __bf16* __restrict__ Bt2,
               int nsrc, float* __restrict__ C, int ldc) {
    constexpr int K = 2048;
    __shared__ __align__(16) __bf16 As[3][128 * PITCH];
    __shared__ __align__(16) __bf16 Bs[3][128 * PITCH];

    const __bf16* Alist[3] = {A0, A1, A2};
    const __bf16* Blist[3] = {Bt0, Bt1, Bt2};

    const int tid  = threadIdx.x;
    const int lane = tid & 31;
    const int wv   = tid >> 5;
    const int wr   = wv & 3;     // wave row   (0..3)  -> 32 rows each
    const int wc   = wv >> 2;    // wave col   (0..1)  -> 64 cols each
    const int l15  = lane & 15;
    const int lh   = lane >> 4;

    const int m0 = blockIdx.y * 128;
    const int n0 = blockIdx.x * 128;

    // staging: each thread async-copies one 32B chunk of A and one of Bt
    const int sm = tid >> 1;          // 0..127 (tile row for A, tile col for B)
    const int sk = (tid & 1) * 16;    // 0 or 16 within BK=32

    const unsigned bufBytes = 128 * PITCH * 2;
    const unsigned laBase   = lds_off(&As[0][0]) + (unsigned)(sm * PITCH + sk) * 2;
    const unsigned lbBase   = lds_off(&Bs[0][0]) + (unsigned)(sm * PITCH + sk) * 2;

    v8f acc[2][4];
#pragma unroll
    for (int i = 0; i < 2; ++i)
#pragma unroll
        for (int j = 0; j < 4; ++j) acc[i][j] = (v8f)(0.0f);

    const int nkt = nsrc * (K / 32);   // always >= 64

    // Async-stage k-tile kt into LDS buffer buf (4 x b128 per thread).
    // NOTE: INST_OFFSET is added to BOTH the LDS and global addresses.
    auto issue_async = [&](int kt, int buf) {
        const int s  = kt >> 6;            // 64 k-tiles per 2048-deep source
        const int kk = (kt & 63) * 32;
        unsigned long long ga =
            (unsigned long long)(Alist[s] + (size_t)(m0 + sm) * K + kk + sk);
        unsigned long long gb =
            (unsigned long long)(Blist[s] + (size_t)(n0 + sm) * K + kk + sk);
        const unsigned la = laBase + (unsigned)buf * bufBytes;
        const unsigned lb = lbBase + (unsigned)buf * bufBytes;
        asm volatile("global_load_async_to_lds_b128 %0, %1, off"
                     :: "v"(la), "v"(ga) : "memory");
        asm volatile("global_load_async_to_lds_b128 %0, %1, off offset:16"
                     :: "v"(la), "v"(ga) : "memory");
        asm volatile("global_load_async_to_lds_b128 %0, %1, off"
                     :: "v"(lb), "v"(gb) : "memory");
        asm volatile("global_load_async_to_lds_b128 %0, %1, off offset:16"
                     :: "v"(lb), "v"(gb) : "memory");
    };

    // Prologue: two tiles in flight, wait for the first (in-order completion).
    issue_async(0, 0);
    issue_async(1, 1);
    asm volatile("s_wait_asynccnt 0x4" ::: "memory");
    __syncthreads();

    for (int kt = 0; kt < nkt; ++kt) {
        const int cur  = kt % 3;
        const bool more = (kt + 1) < nkt;
        const bool pf   = (kt + 2) < nkt;
        if (pf) issue_async(kt + 2, (kt + 2) % 3);   // keep 8 DMAs in flight

        // Fragment loads per CDNA5 16-bit A/B VGPR layout:
        //   A: chunk0 at K = 8*(lane>>4), chunk1 at K = 16 + 8*(lane>>4)
        //   B (N-major LDS): lane n = l15, chunk0 at K = 8*(lane>>4), chunk1 at K = 16 + 8*(lane>>4)
        v16bf af[2], bfrag[4];
#pragma unroll
        for (int i = 0; i < 2; ++i) {
            __bf16* p = &As[cur][(wr * 32 + 16 * i + l15) * PITCH + lh * 8];
            af[i] = mk16(*(const v8bf*)p, *(const v8bf*)(p + 16));
        }
#pragma unroll
        for (int j = 0; j < 4; ++j) {
            __bf16* p = &Bs[cur][(wc * 64 + 16 * j + l15) * PITCH + lh * 8];
            bfrag[j] = mk16(*(const v8bf*)p, *(const v8bf*)(p + 16));
        }
#pragma unroll
        for (int i = 0; i < 2; ++i)
#pragma unroll
            for (int j = 0; j < 4; ++j)
                acc[i][j] = __builtin_amdgcn_wmma_f32_16x16x32_bf16(
                    false, af[i], false, bfrag[j], (short)0, acc[i][j], false, false);

        if (more) {
            // Tile kt+1 (first 4 outstanding DMAs) must be done; tile kt+2 may
            // still be in flight iff it was issued.
            if (pf) asm volatile("s_wait_asynccnt 0x4" ::: "memory");
            else    asm volatile("s_wait_asynccnt 0x0" ::: "memory");
            __syncthreads();
        }
    }

    // C/D layout: lanes 0-15 -> M = e, N = lane; lanes 16-31 -> M = 8+e
#pragma unroll
    for (int i = 0; i < 2; ++i)
#pragma unroll
        for (int j = 0; j < 4; ++j) {
            const int m = m0 + wr * 32 + 16 * i + lh * 8;
            const int n = n0 + wc * 64 + 16 * j + l15;
            float* p = C + (size_t)m * ldc + n;
            if (BETA) {
                float old[8];
#pragma unroll
                for (int e = 0; e < 8; ++e) old[e] = p[(size_t)e * ldc];
#pragma unroll
                for (int e = 0; e < 8; ++e) p[(size_t)e * ldc] = acc[i][j][e] + old[e];
            } else {
#pragma unroll
                for (int e = 0; e < 8; ++e) p[(size_t)e * ldc] = acc[i][j][e];
            }
        }
}

// ---------------------------------------------------------------------------
// Gate nonlinearity: c = sigmoid(f)*c_prev + sigmoid(i)*tanh(g)
// ---------------------------------------------------------------------------
__global__ void __launch_bounds__(256)
gates_kernel(const float* __restrict__ Z, const float* __restrict__ c_prev,
             const float* __restrict__ b_i, const float* __restrict__ b_f,
             const float* __restrict__ b_g,
             float* __restrict__ c_out, __bf16* __restrict__ cbf,
             __bf16* __restrict__ tcbf) {
    size_t idx = (size_t)blockIdx.x * blockDim.x + threadIdx.x;
    if (idx >= ACT) return;
    const int h = (int)(idx & (HH - 1));
    const size_t r = idx >> 11;
    const float* zr = Z + r * LDZ;
    const float iv = zr[h]            + b_i[h];
    const float fv = zr[HH + h]       + b_f[h];
    const float gv = zr[2 * HH + h]   + b_g[h];
    const float c  = sigmoidf_(fv) * c_prev[idx] + sigmoidf_(iv) * tanhf(gv);
    c_out[idx] = c;
    cbf[idx]   = f2bf(c);
    tcbf[idx]  = f2bf(tanhf(c));
}

// ---------------------------------------------------------------------------
// Output: h = tanh(o + b_o) * res   (o slot already += c@w_co, res += tanh(c)@w_c)
// ---------------------------------------------------------------------------
__global__ void __launch_bounds__(256)
hout_kernel(const float* __restrict__ Z, const float* __restrict__ b_o,
            float* __restrict__ h_out) {
    size_t idx = (size_t)blockIdx.x * blockDim.x + threadIdx.x;
    if (idx >= ACT) return;
    const int h = (int)(idx & (HH - 1));
    const size_t r = idx >> 11;
    const float* zr = Z + r * LDZ;
    const float o   = zr[3 * HH + h] + b_o[h];
    const float res = zr[4 * HH + h];
    h_out[idx] = tanhf(o) * res;
}

// ---------------------------------------------------------------------------
// Launch
// ---------------------------------------------------------------------------
extern "C" void kernel_launch(void* const* d_in, const int* in_sizes, int n_in,
                              void* d_out, int out_size, void* d_ws, size_t ws_size,
                              hipStream_t stream) {
    (void)in_sizes; (void)n_in; (void)out_size; (void)ws_size;

    const float* x     = (const float*)d_in[0];
    const float* hprev = (const float*)d_in[1];
    const float* cprev = (const float*)d_in[2];
    const float* W[13];
    for (int j = 0; j < 13; ++j) W[j] = (const float*)d_in[3 + j];
    // W order: 0 w_xi, 1 w_hi, 2 w_ci, 3 w_xf, 4 w_hf, 5 w_cf,
    //          6 w_xg, 7 w_hg, 8 w_xo, 9 w_ho, 10 w_co, 11 w_c, 12 w_i
    const float* b_i = (const float*)d_in[16];
    const float* b_f = (const float*)d_in[17];
    const float* b_g = (const float*)d_in[18];
    const float* b_o = (const float*)d_in[19];

    // workspace layout
    __bf16* wt   = (__bf16*)d_ws;                 // 13 transposed bf16 weights
    __bf16* xbf  = wt + (size_t)13 * NE;
    __bf16* hbf  = xbf + ACT;
    __bf16* cpbf = hbf + ACT;
    float*  Z    = (float*)(cpbf + ACT);          // [B, 5H] fp32 gate buffer
    __bf16* cbf  = (__bf16*)(Z + (size_t)BB * LDZ);
    __bf16* tcbf = cbf + ACT;

    float* h_out = (float*)d_out;                 // tuple (h, c) flat
    float* c_out = h_out + ACT;

    // Phase 1: convert weights (with transpose) and activations to bf16
    dim3 tb(32, 8), tg(64, 64);
    for (int j = 0; j < 13; ++j)
        convT_bf16_kernel<<<tg, tb, 0, stream>>>(W[j], wt + (size_t)j * NE);
    const int n4 = (int)(ACT / 4);
    conv_bf16_kernel<<<(n4 + 255) / 256, 256, 0, stream>>>(x,     xbf,  n4);
    conv_bf16_kernel<<<(n4 + 255) / 256, 256, 0, stream>>>(hprev, hbf,  n4);
    conv_bf16_kernel<<<(n4 + 255) / 256, 256, 0, stream>>>(cprev, cpbf, n4);

    // Phase 2: gate pre-activations, Z[:, g*H : (g+1)*H]
    dim3 gg(2048 / 128, BB / 128);  // (16, 32)
    wmma_gemm_bf16<0><<<gg, 256, 0, stream>>>(xbf, wt + (size_t)0 * NE,
                                              hbf, wt + (size_t)1 * NE,
                                              cpbf, wt + (size_t)2 * NE,
                                              3, Z + 0 * HH, LDZ);   // i
    wmma_gemm_bf16<0><<<gg, 256, 0, stream>>>(xbf, wt + (size_t)3 * NE,
                                              hbf, wt + (size_t)4 * NE,
                                              cpbf, wt + (size_t)5 * NE,
                                              3, Z + 1 * HH, LDZ);   // f
    wmma_gemm_bf16<0><<<gg, 256, 0, stream>>>(xbf, wt + (size_t)6 * NE,
                                              hbf, wt + (size_t)7 * NE,
                                              nullptr, nullptr,
                                              2, Z + 2 * HH, LDZ);   // g
    wmma_gemm_bf16<0><<<gg, 256, 0, stream>>>(xbf, wt + (size_t)8 * NE,
                                              hbf, wt + (size_t)9 * NE,
                                              nullptr, nullptr,
                                              2, Z + 3 * HH, LDZ);   // o (partial)
    wmma_gemm_bf16<0><<<gg, 256, 0, stream>>>(xbf, wt + (size_t)12 * NE,
                                              nullptr, nullptr, nullptr, nullptr,
                                              1, Z + 4 * HH, LDZ);   // residual

    // Phase 3: c = sigmoid(f)*c_prev + sigmoid(i)*tanh(g)
    const int nblk = (int)((ACT + 255) / 256);
    gates_kernel<<<nblk, 256, 0, stream>>>(Z, cprev, b_i, b_f, b_g,
                                           c_out, cbf, tcbf);

    // Phase 4: Z_o += c @ w_co ; Z_res += tanh(c) @ w_c   (accumulate)
    wmma_gemm_bf16<1><<<gg, 256, 0, stream>>>(cbf, wt + (size_t)10 * NE,
                                              nullptr, nullptr, nullptr, nullptr,
                                              1, Z + 3 * HH, LDZ);
    wmma_gemm_bf16<1><<<gg, 256, 0, stream>>>(tcbf, wt + (size_t)11 * NE,
                                              nullptr, nullptr, nullptr, nullptr,
                                              1, Z + 4 * HH, LDZ);

    // Phase 5: h = tanh(o + b_o) * res
    hout_kernel<<<nblk, 256, 0, stream>>>(Z, b_o, h_out);
}
